// QALAS_MAP_57655640981973
// MI455X (gfx1250) — matrix operand order repaired and checked by reading, not compiled, not run on hardware
//
#include <hip/hip_runtime.h>
#include <hip/hip_bf16.h>

typedef __bf16 bf16;
typedef __attribute__((ext_vector_type(16))) __bf16 v16bf;
typedef __attribute__((ext_vector_type(8)))  float  v8f;
typedef int v4i __attribute__((vector_size(16)));   // matches builtin param pointee

#define EPSF 1e-05f
#define DEGF 0.017453292519943295f   // pi/180
#define PI_2F 1.5707963267948966f    // pi/2

#if defined(__gfx1250__) && __has_builtin(__builtin_amdgcn_global_load_async_to_lds_b128)
#define USE_ASYNC_LDS 1
#else
#define USE_ASYNC_LDS 0
#endif

// ---------------- QALAS timing constants (double, as in reference) ----------
constexpr double dESP  = 0.0058;
constexpr double dETL  = 128.0 * 0.0058;
constexpr double dT2P  = 0.1097;
constexpr double dGAP  = 0.9;
constexpr double dTR   = 4.5;
constexpr double dD12  = dT2P;
constexpr double dD01  = dGAP - dETL - dD12;
constexpr double dD23  = dETL;
constexpr double dD45  = 0.0128;
constexpr double dD56  = 0.1 - 0.00645;
constexpr double dD34  = dGAP - dD23 - dD45 - dD56;
constexpr double dD78  = dGAP - dETL;
constexpr double dD910 = dGAP - dETL;
constexpr double dD1112= dGAP - dETL;
constexpr double dTOTAL= dD01 + dD12 + dD23 + dD34 + dD45 + dD56 + dETL + dD78 + dETL
                       + dD910 + dETL + dD1112 + dETL;
constexpr double dD13E = (dTR - dTOTAL) > 0.0 ? (dTR - dTOTAL) : 0.0;

// =============================================================================
// Kernel 1: pack the 5 k-space inputs into NHWC bf16, C padded 5 -> 32
// =============================================================================
__global__ void pack_inputs(const float* __restrict__ a0, const float* __restrict__ a1,
                            const float* __restrict__ a2, const float* __restrict__ a3,
                            const float* __restrict__ a4, bf16* __restrict__ x0, int NP) {
    int p = blockIdx.x * blockDim.x + threadIdx.x;
    if (p >= NP) return;
    bf16* o = x0 + (size_t)p * 32;
    o[0] = (bf16)a0[p];
    o[1] = (bf16)a1[p];
    o[2] = (bf16)a2[p];
    o[3] = (bf16)a3[p];
    o[4] = (bf16)a4[p];
#pragma unroll
    for (int i = 5; i < 32; ++i) o[i] = (bf16)0.0f;
}

// =============================================================================
// Kernel 2: weight reformat  OIHW f32 -> [CoutPad][9*CinPad] bf16
// =============================================================================
__global__ void pack_weights(const float* __restrict__ w, bf16* __restrict__ o,
                             int CinR, int CinP, int CoutR, int CoutP) {
    int ktot = 9 * CinP;
    int total = CoutP * ktot;
    int idx = blockIdx.x * blockDim.x + threadIdx.x;
    if (idx >= total) return;
    int oc  = idx / ktot;
    int k   = idx % ktot;
    int pos = k / CinP;
    int ci  = k % CinP;
    int ky = pos / 3, kx = pos % 3;
    float v = 0.0f;
    if (oc < CoutR && ci < CinR)
        v = w[(((size_t)oc * CinR + ci) * 3 + ky) * 3 + kx];
    o[idx] = (bf16)v;
}

// =============================================================================
// Kernel 3: implicit-GEMM 3x3 conv + bias + ReLU via v_wmma_f32_16x16x32_bf16
//   - whole 3-row halo staged in LDS ONCE (async-to-LDS when available)
//   - fully unrolled K loop: 9*(CINP/32)*TPW back-to-back WMMAs
//   - epilogue through LDS -> coalesced b128 stores
// =============================================================================
template<int CINP, int COUTP, int COUT_REAL, int TPW>
__global__ __launch_bounds__(256)
void conv_wmma(const bf16* __restrict__ act, const bf16* __restrict__ wgt,
               const float* __restrict__ bias, bf16* __restrict__ outp,
               int H, int W) {
    constexpr int KTOT = 9 * CINP;
    constexpr int CS   = CINP / 32;        // K sub-steps per (ky,kx)
    constexpr int LDP  = CINP + 8;         // LDS pixel stride (bf16), 16B-aligned
    constexpr int LDO  = COUTP + 8;        // LDS output stride (bf16), 16B-aligned
    __shared__ bf16 smA[3 * 66 * LDP];
    __shared__ bf16 smO[64 * LDO];

    const int tid   = threadIdx.x;
    const int lane  = tid & 31;
    const int wave  = tid >> 5;
    const int nthr  = blockDim.x;
    const int x0    = blockIdx.x * 64;
    const int y     = blockIdx.y;
    const int b     = blockIdx.z;
    const int laneM = lane & 15;
    const int kh8   = (lane >> 4) * 8;     // A-frag K-half offset
    const int kh16  = (lane >> 4) * 8;     // B-frag K-half offset
    const int ni    = (wave * TPW) >> 2;   // all TPW tiles of a wave share ni

    const int  nB    = ni * 16 + laneM;    // weight column for this lane
    const bf16* wrow = wgt + (size_t)nB * KTOT + kh16;
    __builtin_prefetch((const void*)wrow, 0, 1);   // global_prefetch_b8

    // ---- stage full 3x66xCINP bf16 halo into LDS (once) ----
    constexpr int CPP    = CINP / 8;           // 16B chunks per pixel
    constexpr int CHUNKS = 3 * 66 * CPP;
    for (int c = tid; c < CHUNKS; c += nthr) {
        const int ko  = (c % CPP) * 8;
        const int pix = c / CPP;
        const int r   = pix / 66;              // 0..2  ->  y-1..y+1
        const int px  = pix % 66;              // 0..65 ->  x0-1..x0+64
        const int yy  = y + r - 1;
        const int xx  = x0 + px - 1;
        bf16* dst = &smA[(r * 66 + px) * LDP + ko];
        if (yy >= 0 && yy < H && xx >= 0 && xx < W) {
            const bf16* src = act + (((size_t)b * H + yy) * W + xx) * CINP + ko;
#if USE_ASYNC_LDS
            __builtin_amdgcn_global_load_async_to_lds_b128(
                (__attribute__((address_space(1))) v4i*)src,
                (__attribute__((address_space(3))) v4i*)dst, 0, 0);
#else
            *(int4*)dst = *(const int4*)src;
#endif
        } else {
            *(int4*)dst = make_int4(0, 0, 0, 0);
        }
    }
#if USE_ASYNC_LDS
#if __has_builtin(__builtin_amdgcn_s_wait_asynccnt)
    __builtin_amdgcn_s_wait_asynccnt(0);
#else
    asm volatile("s_wait_asynccnt 0" ::: "memory");
#endif
#endif
    __syncthreads();

    // ---- fully unrolled K loop: 9 taps x CS sub-steps ----
    v8f acc[TPW];
#pragma unroll
    for (int t = 0; t < TPW; ++t)
#pragma unroll
        for (int r = 0; r < 8; ++r) acc[t][r] = 0.0f;

#pragma unroll
    for (int pos = 0; pos < 9; ++pos) {
        const int ky = pos / 3;                // LDS row 0..2
        const int kx = pos % 3;                // LDS px offset 0..2 (== kx+1)
#pragma unroll
        for (int cs = 0; cs < CS; ++cs) {
            const int ks = pos * CS + cs;
            union { v16bf v; int4 q[2]; } bf_;
            bf_.q[0] = *(const int4*)(wrow + ks * 32);
            bf_.q[1] = *(const int4*)(wrow + ks * 32 + 16);
#pragma unroll
            for (int t = 0; t < TPW; ++t) {
                const int mi = (wave * TPW + t) & 3;
                const bf16* ap = &smA[(ky * 66 + mi * 16 + laneM + kx) * LDP
                                      + cs * 32 + kh8];
                union { v16bf v; int4 q[2]; } af;
                af.q[0] = *(const int4*)ap;
                af.q[1] = *(const int4*)(ap + 16);
                acc[t] = __builtin_amdgcn_wmma_f32_16x16x32_bf16(
                             false, af.v, false, bf_.v, (short)0, acc[t], false, false);
            }
        }
    }

    // ---- epilogue: bias + ReLU -> LDS tile -> coalesced b128 stores ----
    const int   ch = ni * 16 + laneM;
    const float bv = (ch < COUT_REAL) ? bias[ch] : 0.0f;
#pragma unroll
    for (int t = 0; t < TPW; ++t) {
        const int mi = (wave * TPW + t) & 3;
#pragma unroll
        for (int r = 0; r < 8; ++r) {
            const int px = mi * 16 + r + ((lane >> 4) * 8);
            float v = acc[t][r] + bv;
            v = v > 0.0f ? v : 0.0f;
            smO[px * LDO + ch] = (bf16)v;
        }
    }
    __syncthreads();

    constexpr int OCH = COUTP / 8;             // 16B chunks per pixel
    for (int c = tid; c < 64 * OCH; c += nthr) {
        const int px = c / OCH;
        const int ko = (c % OCH) * 8;
        const int x  = x0 + px;
        *(int4*)(outp + (((size_t)b * H + y) * W + x) * COUTP + ko) =
            *(const int4*)&smO[px * LDO + ko];
    }
}

// =============================================================================
// Kernel 4: per-pixel QALAS signal equation (20-iteration relaxation recursion)
// =============================================================================
__device__ __forceinline__ float relaxf(float m, float E, float m0) {
    return fmaf(m, E, m0 * (1.0f - E));
}

__global__ void qalas_kernel(const bf16* __restrict__ fin, const float* __restrict__ b1,
                             const float* __restrict__ mv_t1, const float* __restrict__ mv_t2,
                             float* __restrict__ out, int NP) {
    int p = blockIdx.x * blockDim.x + threadIdx.x;
    if (p >= NP) return;

    const float c0 = (float)fin[(size_t)p * 16 + 0];
    const float c1 = (float)fin[(size_t)p * 16 + 1];
    const float c2 = (float)fin[(size_t)p * 16 + 2];
    const float c3 = (float)fin[(size_t)p * 16 + 3];

    const float t1  = c0 * mv_t1[0];
    const float t2  = c1 * mv_t2[0];
    const float pd  = c2 / sinf(DEGF * 4.0f);
    const float iev = c3 * 0.5f + 0.5f;
    const float b1v = b1[p];

    const float inv_t1 = 1.0f / (t1 + EPSF);
    const float ET2  = expf(-((float)dD12 - 0.0097f) / (t2 + EPSF));
    const float ET1  = expf(-((float)dD12 - 0.0097f) * inv_t1);
    const float Ed1  = expf(-(float)dD01  * inv_t1);
    const float Ed4  = expf(-(float)dD34  * inv_t1);
    const float Ed6  = expf(-(float)dD56  * inv_t1);
    const float Ed8  = expf(-(float)dD78  * inv_t1);
    const float Ed10 = expf(-(float)dD910 * inv_t1);
    const float Ed12 = expf(-(float)dD1112* inv_t1);
    const float Ed14 = expf(-(float)dD13E * inv_t1);
    const float Eda  = expf(-0.0097f * inv_t1);
    const float Edb  = 1.0f;                       // exp(-0/t1)

    const float flip    = 4.0f * b1v;
    const float t1_star = t1 / (1.0f - t1 * logf(cosf(DEGF * flip)) / (float)dESP);
    const float m0_star = pd * (1.0f - expf(-(float)dESP * inv_t1))
                             / (1.0f - expf(-(float)dESP / (t1_star + EPSF)));
    const float Eetl   = expf(-(float)dETL / (t1_star + EPSF));
    const float sin_fa = sinf(DEGF * flip);
    const float t2r    = PI_2F * b1v;
    const float s = sinf(t2r), c = cosf(t2r);
    const float prep = s * s * ET2 + c * c * ET1;

    float m = pd, a1 = pd, a2 = pd, a3 = pd, a4 = pd, a5 = pd;
#pragma unroll 1
    for (int it = 0; it < 20; ++it) {
        m = relaxf(m, Ed1, pd);
        m = m * prep;
        m = relaxf(m, Eda, pd);
        a1 = m * sin_fa;
        m = relaxf(m, Eetl, m0_star);
        m = relaxf(m, Ed4, pd);
        m = -m * iev;
        m = relaxf(m, Ed6, pd);
        m = relaxf(m, Edb, pd);
        a2 = m * sin_fa;
        m = relaxf(m, Eetl, m0_star);
        m = relaxf(m, Ed8, pd);
        m = relaxf(m, Edb, pd);
        a3 = m * sin_fa;
        m = relaxf(m, Eetl, m0_star);
        m = relaxf(m, Ed10, pd);
        m = relaxf(m, Edb, pd);
        a4 = m * sin_fa;
        m = relaxf(m, Eetl, m0_star);
        m = relaxf(m, Ed12, pd);
        m = relaxf(m, Edb, pd);
        a5 = m * sin_fa;
        m = relaxf(m, Eetl, m0_star);
        m = relaxf(m, Ed14, pd);
    }

    out[0 * (size_t)NP + p] = t1;
    out[1 * (size_t)NP + p] = t2;
    out[2 * (size_t)NP + p] = pd;
    out[3 * (size_t)NP + p] = iev;
    out[4 * (size_t)NP + p] = b1v;
    out[5 * (size_t)NP + p] = a1;
    out[6 * (size_t)NP + p] = a2;
    out[7 * (size_t)NP + p] = a3;
    out[8 * (size_t)NP + p] = a4;
    out[9 * (size_t)NP + p] = a5;
}

// =============================================================================
// Host launcher
// =============================================================================
extern "C" void kernel_launch(void* const* d_in, const int* in_sizes, int n_in,
                              void* d_out, int out_size, void* d_ws, size_t ws_size,
                              hipStream_t stream) {
    constexpr int B = 8, H = 320, W = 320;
    const size_t NP = (size_t)B * H * W;

    const float* ksp[5];
    for (int i = 0; i < 5; ++i) ksp[i] = (const float*)d_in[i];
    const float* b1    = (const float*)d_in[11];
    const float* mv_t1 = (const float*)d_in[13];
    const float* mv_t2 = (const float*)d_in[14];
    const float* cw[5];
    const float* cb[5];
    for (int i = 0; i < 5; ++i) {
        cw[i] = (const float*)d_in[17 + 2 * i];
        cb[i] = (const float*)d_in[18 + 2 * i];
    }

    char* ws = (char*)d_ws;
    size_t off = 0;
    auto carve = [&](size_t bytes) -> void* {
        off = (off + 255) & ~(size_t)255;
        void* p = ws + off;
        off += bytes;
        return p;
    };
    bf16* x0   = (bf16*)carve(NP * 32 * sizeof(bf16));
    bf16* actA = (bf16*)carve(NP * 64 * sizeof(bf16));
    bf16* actB = (bf16*)carve(NP * 64 * sizeof(bf16));
    bf16* fin  = (bf16*)carve(NP * 16 * sizeof(bf16));
    bf16* w0   = (bf16*)carve((size_t)64 * 9 * 32 * sizeof(bf16));
    bf16* w1   = (bf16*)carve((size_t)64 * 9 * 64 * sizeof(bf16));
    bf16* w2   = (bf16*)carve((size_t)64 * 9 * 64 * sizeof(bf16));
    bf16* w3   = (bf16*)carve((size_t)64 * 9 * 64 * sizeof(bf16));
    bf16* w4   = (bf16*)carve((size_t)16 * 9 * 64 * sizeof(bf16));

    pack_inputs<<<(int)((NP + 255) / 256), 256, 0, stream>>>(
        ksp[0], ksp[1], ksp[2], ksp[3], ksp[4], x0, (int)NP);

    auto wgrid = [](int total) { return dim3((total + 255) / 256); };
    pack_weights<<<wgrid(64 * 9 * 32), 256, 0, stream>>>(cw[0], w0,  5, 32, 64, 64);
    pack_weights<<<wgrid(64 * 9 * 64), 256, 0, stream>>>(cw[1], w1, 64, 64, 64, 64);
    pack_weights<<<wgrid(64 * 9 * 64), 256, 0, stream>>>(cw[2], w2, 64, 64, 64, 64);
    pack_weights<<<wgrid(64 * 9 * 64), 256, 0, stream>>>(cw[3], w3, 64, 64, 64, 64);
    pack_weights<<<wgrid(16 * 9 * 64), 256, 0, stream>>>(cw[4], w4, 64, 64,  4, 16);

    dim3 cgrid(W / 64, H, B);
    conv_wmma<32, 64, 64, 2><<<cgrid, 256, 0, stream>>>(x0,   w0, cb[0], actA, H, W);
    conv_wmma<64, 64, 64, 2><<<cgrid, 256, 0, stream>>>(actA, w1, cb[1], actB, H, W);
    conv_wmma<64, 64, 64, 2><<<cgrid, 256, 0, stream>>>(actB, w2, cb[2], actA, H, W);
    conv_wmma<64, 64, 64, 2><<<cgrid, 256, 0, stream>>>(actA, w3, cb[3], actB, H, W);
    conv_wmma<64, 16,  4, 1><<<cgrid, 128, 0, stream>>>(actB, w4, cb[4], fin,  H, W);

    qalas_kernel<<<(int)((NP + 255) / 256), 256, 0, stream>>>(
        fin, b1, mv_t1, mv_t2, (float*)d_out, (int)NP);
}
